// TopkRouter_5686536700608
// MI455X (gfx1250) — matrix operand order repaired and checked by reading, not compiled, not run on hardware
//
#include <hip/hip_runtime.h>
#include <stdint.h>

typedef __attribute__((ext_vector_type(16))) _Float16 v16h;
typedef __attribute__((ext_vector_type(8)))  _Float16 v8h;
typedef __attribute__((ext_vector_type(8)))  float    v8f;
typedef __attribute__((ext_vector_type(4)))  float    v4f;
typedef __attribute__((ext_vector_type(4)))  uint32_t u32x4;
typedef __attribute__((ext_vector_type(8)))  int      i32x8;
typedef __attribute__((ext_vector_type(4)))  int      i32x4;

#define EMBED  2048
#define NEXP   64
#define NTOK   16384          // 4 * 4096
#define TOPK   8
#define KCHUNK 128
#define NCHUNK (EMBED / KCHUNK)   // 16
#define LSTR   (KCHUNK + 8)       // 136-half padded LDS row stride (bank-conflict-free)

#define WMMA16(A, B, C) \
    __builtin_amdgcn_wmma_f32_16x16x32_f16(false, (A), false, (B), (short)0, (C), false, false)

#define LOADB(dst, rowofs, kofs) do { \
    v8h t0_ = *(const v8h*)(lb + ((rowofs) + r) * LSTR + (kofs)); \
    v8h t1_ = *(const v8h*)(lb + ((rowofs) + r) * LSTR + (kofs) + 16); \
    dst = __builtin_shufflevector(t0_, t1_, 0,1,2,3,4,5,6,7,8,9,10,11,12,13,14,15); \
} while (0)

#if __has_builtin(__builtin_amdgcn_sched_group_barrier)
#define SGB(mask, cnt, id) __builtin_amdgcn_sched_group_barrier((mask), (cnt), (id))
#else
#define SGB(mask, cnt, id)
#endif
// sched_group_barrier masks: 0x020 = VMEM read, 0x100 = DS read, 0x008 = MFMA/WMMA

// ---------------------------------------------------------------------------
// Kernel 0: W (64 x 2048 f32) -> f16, once.
// ---------------------------------------------------------------------------
__global__ void __launch_bounds__(256) cvt_w_kernel(const float* __restrict__ W,
                                                    _Float16* __restrict__ Wh) {
    int i = blockIdx.x * 256 + threadIdx.x;
    Wh[i] = (_Float16)W[i];
}

// ---------------------------------------------------------------------------
// TDM: DMA one W K-chunk (64 rows x 128 f16) global -> LDS, padding 4 DWORDs
// after every 64 DWORDs (one 128-half row) to produce the 136-half stride.
// ---------------------------------------------------------------------------
__device__ __forceinline__ void tdm_load_w_chunk(const _Float16* src, uint32_t lds_off) {
    const unsigned long long ga = (unsigned long long)(uintptr_t)src;
    u32x4 g0;
    g0[0] = 1u;                                   // count=1, user mode, no gather
    g0[1] = lds_off;                              // lds_addr (bytes)
    g0[2] = (uint32_t)ga;                         // global_addr[31:0]
    g0[3] = (uint32_t)((ga >> 32) & 0x01FFFFFFu)  // global_addr[56:32]
          | (2u << 30);                           // type=2 ("image")
    i32x8 g1;
    g1[0] = (1 << 16)        // data_size = 1 -> 2 bytes
          | (1 << 20)        // pad_enable
          | (5 << 22)        // pad_interval: every 64 DWORDs (256B row)
          | (3 << 25);       // pad_amount:   4 DWORDs (16B = 8 halves)
    g1[1] = (int)((EMBED & 0xFFFFu) << 16);                  // tensor_dim0[15:0]
    g1[2] = (int)(((EMBED >> 16) & 0xFFFFu) | (NEXP << 16)); // tensor_dim0 hi | tensor_dim1 lo
    g1[3] = (int)(KCHUNK << 16);                             // tile_dim0 = 128
    g1[4] = NEXP;                                            // tile_dim1 = 64 rows
    g1[5] = EMBED;                                           // tensor_dim0_stride = 2048
    g1[6] = 0;
    g1[7] = 0;
    i32x4 gz = {0, 0, 0, 0};
#if __clang_major__ >= 23
    i32x8 gz8 = {0, 0, 0, 0, 0, 0, 0, 0};
    __builtin_amdgcn_tensor_load_to_lds(g0, g1, gz, gz, gz8, 0);
#else
    __builtin_amdgcn_tensor_load_to_lds(g0, g1, gz, gz, 0);
#endif
}

// ---------------------------------------------------------------------------
// Kernel 1: logits = X * W^T + b.
//   - 8 waves/block, 1 wave = 16 tokens x 64 experts (4 WMMA accumulators)
//   - W staged into double-buffered LDS by the Tensor Data Mover (1 DMA/chunk)
//   - B fragments 1 k-step ahead (pinned via sched_group_barrier)
//   - X stream 2 k-steps ahead (HBM-critical)
// ---------------------------------------------------------------------------
__global__ void __launch_bounds__(256) router_gemm_kernel(
    const float*    __restrict__ X,
    const _Float16* __restrict__ Wh,
    const float*    __restrict__ bias,
    float*          __restrict__ logits)
{
    __shared__ _Float16 wbuf[2][NEXP * LSTR];   // 2 * 64 * 136 * 2B = 34816 B

    const int tid     = threadIdx.x;
    const int lane    = tid & 31;
    const int wave    = tid >> 5;
    const int tokBase = blockIdx.x * 128 + wave * 16;
    const int r       = lane & 15;   // M row / N col within 16x16 tile
    const int hi      = lane >> 4;   // half-wave selector
    const int kA      = hi * 8;      // A frag K offset (ISA 7.12.2)
    const int kB      = hi * 8;      // B frag K offset

    const float* aptr = X + (size_t)(tokBase + r) * EMBED;

    v8f acc0 = {}, acc1 = {}, acc2 = {}, acc3 = {};

    // ---- prologue: TDM chunk 0 into LDS buffer 0 ----
    if (wave == 0) {
        tdm_load_w_chunk(Wh, (uint32_t)(uintptr_t)&wbuf[0][0]);
    }
    __builtin_amdgcn_s_wait_tensorcnt(0);
    __syncthreads();

    // ---- A raw fragments for k=0 and k=32 (depth-2 pipeline) ----
    v4f a0 = *(const v4f*)(aptr + kA);
    v4f a1 = *(const v4f*)(aptr + kA + 4);
    v4f a2 = *(const v4f*)(aptr + kA + 16);
    v4f a3 = *(const v4f*)(aptr + kA + 20);
    v4f x0 = *(const v4f*)(aptr + 32 + kA);
    v4f x1 = *(const v4f*)(aptr + 32 + kA + 4);
    v4f x2 = *(const v4f*)(aptr + 32 + kA + 16);
    v4f x3 = *(const v4f*)(aptr + 32 + kA + 20);

    for (int c = 0; c < NCHUNK; ++c) {
        const int bsel = c & 1;
        const _Float16* lb = &wbuf[bsel][0];
        const bool has_next = (c + 1) < NCHUNK;

        // kick off next chunk's W DMA; runs concurrent with this chunk's math
        if (has_next && wave == 0) {
            tdm_load_w_chunk(Wh + (c + 1) * KCHUNK,
                             (uint32_t)(uintptr_t)&wbuf[bsel ^ 1][0]);
        }

        // B fragments for step 0 of this chunk
        v16h bc0, bc1, bc2, bc3, bn0, bn1, bn2, bn3;
        LOADB(bc0, 0,  kB); LOADB(bc1, 16, kB);
        LOADB(bc2, 32, kB); LOADB(bc3, 48, kB);
        SGB(0x100, 8, 0);                 // group: these 8 DS reads first

#pragma unroll
        for (int s = 0; s < 4; ++s) {
            const int k  = c * KCHUNK + s * 32;
            const int kn = (k + 64) & (EMBED - 1);   // depth-2; wrap stays in-bounds

            // prefetch A raw data two k-steps ahead
            v4f t0 = *(const v4f*)(aptr + kn + kA);
            v4f t1 = *(const v4f*)(aptr + kn + kA + 4);
            v4f t2 = *(const v4f*)(aptr + kn + kA + 16);
            v4f t3 = *(const v4f*)(aptr + kn + kA + 20);

            // prefetch next k-step's B fragments from LDS
            if (s < 3) {
                const int kofn = (s + 1) * 32 + kB;
                LOADB(bn0, 0,  kofn); LOADB(bn1, 16, kofn);
                LOADB(bn2, 32, kofn); LOADB(bn3, 48, kofn);
            }

            // convert current A raw -> f16 fragment
            v16h af;
#pragma unroll
            for (int q = 0; q < 4; ++q) {
                af[q]      = (_Float16)a0[q];
                af[4 + q]  = (_Float16)a1[q];
                af[8 + q]  = (_Float16)a2[q];
                af[12 + q] = (_Float16)a3[q];
            }

            acc0 = WMMA16(af, bc0, acc0);
            acc1 = WMMA16(af, bc1, acc1);
            acc2 = WMMA16(af, bc2, acc2);
            acc3 = WMMA16(af, bc3, acc3);

            // pin issue order for this step: A vmem, next-B ds, then WMMAs
            SGB(0x020, 4, 0);             // 4 VMEM reads (A prefetch)
            if (s < 3) SGB(0x100, 8, 0);  // 8 DS reads (next-step B)
            SGB(0x008, 4, 0);             // 4 WMMA

            if (s < 3) { bc0 = bn0; bc1 = bn1; bc2 = bn2; bc3 = bn3; }
            a0 = x0; a1 = x1; a2 = x2; a3 = x3;
            x0 = t0; x1 = t1; x2 = t2; x3 = t3;
        }

        // DMA for chunk c+1 must have landed before next chunk reads it
        __builtin_amdgcn_s_wait_tensorcnt(0);
        __syncthreads();
    }

    // epilogue: bias add + store per C layout (VGPR j -> M=j lanes<16, M=j+8 lanes>=16)
    const float bb0 = bias[r], bb1 = bias[16 + r], bb2 = bias[32 + r], bb3 = bias[48 + r];
    const int mo = hi * 8;
#pragma unroll
    for (int j = 0; j < 8; ++j) {
        float* op = logits + (size_t)(tokBase + mo + j) * NEXP + r;
        op[0]  = acc0[j] + bb0;
        op[16] = acc1[j] + bb1;
        op[32] = acc2[j] + bb2;
        op[48] = acc3[j] + bb3;
    }
}

// ---------------------------------------------------------------------------
// Kernel 2: per-token softmax -> top-8 -> masked softmax. One thread/token;
// scores fully register-resident; sel bitmask avoids dynamic array writes.
// ---------------------------------------------------------------------------
__global__ void __launch_bounds__(256) router_topk_kernel(
    const float* __restrict__ logits,
    float*       __restrict__ out_router,
    float*       __restrict__ out_mask)
{
    const int t = blockIdx.x * 256 + threadIdx.x;
    const float* lp = logits + (size_t)t * NEXP;

    float sc[NEXP];
#pragma unroll
    for (int i = 0; i < NEXP; i += 4) {
        v4f v = *(const v4f*)(lp + i);
        sc[i] = v[0]; sc[i+1] = v[1]; sc[i+2] = v[2]; sc[i+3] = v[3];
    }

    // softmax over all 64 experts
    float m = sc[0];
#pragma unroll
    for (int i = 1; i < NEXP; ++i) m = fmaxf(m, sc[i]);
    float s = 0.0f;
#pragma unroll
    for (int i = 0; i < NEXP; ++i) { sc[i] = __expf(sc[i] - m); s += sc[i]; }
    const float inv = 1.0f / s;
#pragma unroll
    for (int i = 0; i < NEXP; ++i) sc[i] *= inv;

    // top-8 via bitmask argmax passes (scores in [0,1])
    unsigned long long sel = 0ull;
#pragma unroll 1
    for (int p = 0; p < TOPK; ++p) {
        float mv = -1.0f;
        int   mi = 0;
#pragma unroll
        for (int i = 0; i < NEXP; ++i) {
            const bool  taken = (sel >> i) & 1ull;
            const float v     = taken ? -1.0f : sc[i];
            if (v > mv) { mv = v; mi = i; }
        }
        sel |= (1ull << mi);
    }

    // masked re-softmax over selected experts
    float z = 0.0f;
#pragma unroll
    for (int i = 0; i < NEXP; ++i) {
        const bool on = (sel >> i) & 1ull;
        sc[i] = on ? __expf(sc[i]) : 0.0f;
        z += sc[i];
    }
    const float invz = 1.0f / z;

    float* rp = out_router + (size_t)t * NEXP;
    float* mp = out_mask   + (size_t)t * NEXP;
#pragma unroll
    for (int i = 0; i < NEXP; i += 4) {
        v4f rv, mv4;
#pragma unroll
        for (int j = 0; j < 4; ++j) {
            const bool on = (sel >> (i + j)) & 1ull;
            rv[j]  = sc[i + j] * invz;
            mv4[j] = on ? 1.0f : 0.0f;
        }
        *(v4f*)(rp + i) = rv;
        *(v4f*)(mp + i) = mv4;
    }
}

// ---------------------------------------------------------------------------
extern "C" void kernel_launch(void* const* d_in, const int* in_sizes, int n_in,
                              void* d_out, int out_size, void* d_ws, size_t ws_size,
                              hipStream_t stream) {
    const float* X = (const float*)d_in[0];   // (4,4096,2048) f32
    const float* W = (const float*)d_in[1];   // (64,2048)     f32
    const float* b = (const float*)d_in[2];   // (64,)         f32

    float* out_router = (float*)d_out;                          // (16384,64)
    float* out_mask   = (float*)d_out + (size_t)NTOK * NEXP;    // (16384,64)

    _Float16* Wh     = (_Float16*)d_ws;                                   // 256 KB
    float*    logits = (float*)((char*)d_ws + (size_t)NEXP * EMBED * 2);  // 4 MB

    cvt_w_kernel<<<(NEXP * EMBED) / 256, 256, 0, stream>>>(W, Wh);
    router_gemm_kernel<<<NTOK / 128, 256, 0, stream>>>(X, Wh, b, logits);
    router_topk_kernel<<<NTOK / 256, 256, 0, stream>>>(logits, out_router, out_mask);
}
